// AdditiveMask_9096740733412
// MI455X (gfx1250) — matrix operand order repaired and checked by reading, not compiled, not run on hardware
//
#include <hip/hip_runtime.h>
#include <hip/hip_bf16.h>
#include <math.h>

typedef __attribute__((ext_vector_type(16))) _Float16 v16h;
typedef __attribute__((ext_vector_type(8)))  float    v8f;

#define DIM 128
#define EPS 1e-8f

// ---------------------------------------------------------------------------
// Kernel 1: h = tanh(x @ W^T + b), tiled with v_wmma_f32_16x16x32_f16.
// Block = 256 threads (8 wave32s). Block handles a 16-node tile x all 128 dims;
// wave w handles output-dim tile [16w, 16w+16).
// ---------------------------------------------------------------------------
__global__ void __launch_bounds__(256)
gemm_tanh_wmma(const float* __restrict__ x, const float* __restrict__ W,
               const float* __restrict__ bias, float* __restrict__ h, int nNodes) {
  __shared__ _Float16 Wh[DIM * DIM];   // 32 KB: Wh[d][k] = (f16)W[d][k]
  __shared__ _Float16 Xh[16 * DIM];    //  4 KB: 16-node tile of x

  const int tid      = threadIdx.x;
  const int nodeBase = blockIdx.x * 16;

  // Cooperative stage of W (f32 -> f16) into LDS
  for (int i = tid; i < DIM * DIM; i += 256) Wh[i] = (_Float16)W[i];
  // Cooperative stage of the x tile
  for (int i = tid; i < 16 * DIM; i += 256) {
    int r = i >> 7, k = i & (DIM - 1);
    Xh[i] = (_Float16)x[(size_t)(nodeBase + r) * DIM + k];
  }
  __syncthreads();

  const int wave    = tid >> 5;
  const int lane    = tid & 31;
  const int half    = lane >> 4;        // which 16-lane half
  const int rowA    = lane & 15;        // A-matrix row (M)
  const int dimBase = wave * 16;
  const int colB    = dimBase + (lane & 15);  // B-matrix column (N)

  v8f acc = {};
  #pragma unroll
  for (int k0 = 0; k0 < DIM; k0 += 32) {
    // A fragment (16x32 f16): lane half selects K offset {0,8}, VGPRs 4-7 add 16
    const int ka = k0 + half * 8;
    v16h a, bm;
    #pragma unroll
    for (int e = 0; e < 16; ++e) {
      int k = ka + e + ((e >= 8) ? 8 : 0);
      a[e] = Xh[rowA * DIM + k];
    }
    // B fragment (32x16 f16): B[k][n] = W[n][k]; lane half selects K offset {0,8}, VGPRs 4-7 add 16
    const int kb = k0 + half * 8;
    #pragma unroll
    for (int e = 0; e < 16; ++e) bm[e] = Wh[colB * DIM + kb + e + ((e >= 8) ? 8 : 0)];

    acc = __builtin_amdgcn_wmma_f32_16x16x32_f16(
        /*neg_a=*/false, a, /*neg_b=*/false, bm,
        /*c_mod=*/(short)0, acc, /*reuse_a=*/false, /*reuse_b=*/false);
  }

  // C/D layout: VGPR r -> M = r + 8*half, N = lane&15
  const float bv = bias[colB];
  #pragma unroll
  for (int r = 0; r < 8; ++r) {
    int m = r + half * 8;
    h[(size_t)(nodeBase + m) * DIM + colB] = tanhf(acc[r] + bv);
  }
}

// ---------------------------------------------------------------------------
// Kernel 2: per-node inverse norm rinv[n] = 1/max(||h_n||, EPS); also deg=1.0
// (self-loop weight). One wave32 per node, float4 per lane.
// ---------------------------------------------------------------------------
__global__ void __launch_bounds__(256)
node_rinv(const float* __restrict__ h, float* __restrict__ rinv,
          float* __restrict__ deg, int nNodes) {
  const int n    = blockIdx.x * 8 + (threadIdx.x >> 5);
  const int lane = threadIdx.x & 31;
  if (n >= nNodes) return;
  const float4 v = *(const float4*)(h + (size_t)n * DIM + lane * 4);
  float s = v.x * v.x + v.y * v.y + v.z * v.z + v.w * v.w;
  #pragma unroll
  for (int m = 16; m >= 1; m >>= 1) s += __shfl_xor(s, m, 32);
  if (lane == 0) {
    rinv[n] = 1.0f / fmaxf(sqrtf(s), EPS);
    deg[n]  = 1.0f;   // self-loop contribution to degree
  }
}

// ---------------------------------------------------------------------------
// Kernel 3: edge weights ew[e] = relu( dot(h[s],h[d]) * rinv[s] * rinv[d] ).
// One wave32 per edge; h rows are L2-resident (51MB < 192MB L2).
// ---------------------------------------------------------------------------
__global__ void __launch_bounds__(256)
edge_weights(const float* __restrict__ h, const float* __restrict__ rinv,
             const int* __restrict__ ei, float* __restrict__ ew, int nEdges) {
  const int e    = blockIdx.x * 8 + (threadIdx.x >> 5);
  const int lane = threadIdx.x & 31;
  if (e >= nEdges) return;
  const int s = ei[e];
  const int d = ei[nEdges + e];
  const float4 a = *(const float4*)(h + (size_t)s * DIM + lane * 4);
  const float4 b = *(const float4*)(h + (size_t)d * DIM + lane * 4);
  float dot = a.x * b.x + a.y * b.y + a.z * b.z + a.w * b.w;
  #pragma unroll
  for (int m = 16; m >= 1; m >>= 1) dot += __shfl_xor(dot, m, 32);
  if (lane == 0) ew[e] = fmaxf(dot * rinv[s] * rinv[d], 0.0f);
}

// ---------------------------------------------------------------------------
// Kernel 4: degree accumulation over dst (segment_sum of w over col).
// ---------------------------------------------------------------------------
__global__ void __launch_bounds__(256)
deg_accum(const int* __restrict__ ei, const float* __restrict__ ew,
          float* __restrict__ deg, int nEdges) {
  const int e = blockIdx.x * 256 + threadIdx.x;
  if (e >= nEdges) return;
  atomicAdd(deg + ei[nEdges + e], ew[e]);
}

// ---------------------------------------------------------------------------
// Kernel 5: dinv[n] = deg>0 ? rsqrt(max(deg,EPS)) : 0 (in-place over deg);
// also init z = relu(mask) and agg = dinv^2 * z (self-loop term of step 1).
// ---------------------------------------------------------------------------
__global__ void __launch_bounds__(256)
dinv_init(float* __restrict__ deg, const float* __restrict__ mask,
          float* __restrict__ z, float* __restrict__ agg, int nNodes) {
  const int n = blockIdx.x * 256 + threadIdx.x;
  if (n >= nNodes) return;
  const float dg = deg[n];
  const float di = (dg > 0.0f) ? rsqrtf(fmaxf(dg, EPS)) : 0.0f;
  deg[n] = di;                         // deg array now holds dinv
  const float fv = fmaxf(mask[n], 0.0f);
  z[n]   = fv;
  agg[n] = di * di * fv;
}

// ---------------------------------------------------------------------------
// Kernel 6: per-edge propagation coefficient enorm = dinv[src]*ew*dinv[dst].
// ---------------------------------------------------------------------------
__global__ void __launch_bounds__(256)
edge_norm(const int* __restrict__ ei, const float* __restrict__ ew,
          const float* __restrict__ dinv, float* __restrict__ enorm, int nEdges) {
  const int e = blockIdx.x * 256 + threadIdx.x;
  if (e >= nEdges) return;
  enorm[e] = dinv[ei[e]] * ew[e] * dinv[ei[nEdges + e]];
}

// ---------------------------------------------------------------------------
// Kernel 7: one APPNP scatter step: agg[dst] += enorm[e] * z[src].
// ---------------------------------------------------------------------------
__global__ void __launch_bounds__(256)
appnp_scatter(const int* __restrict__ ei, const float* __restrict__ enorm,
              const float* __restrict__ z, float* __restrict__ agg, int nEdges) {
  const int e = blockIdx.x * 256 + threadIdx.x;
  if (e >= nEdges) return;
  atomicAdd(agg + ei[nEdges + e], enorm[e] * z[ei[e]]);
}

// ---------------------------------------------------------------------------
// Kernel 8: z = (1-a)*agg + a*fill; re-init agg = dinv^2 * z for next step.
// ---------------------------------------------------------------------------
__global__ void __launch_bounds__(256)
appnp_update(const float* __restrict__ mask, const float* __restrict__ dinv,
             const float* __restrict__ alpha, float* __restrict__ z,
             float* __restrict__ agg, int nNodes) {
  const int n = blockIdx.x * 256 + threadIdx.x;
  if (n >= nNodes) return;
  const float al = alpha[0];
  const float fv = fmaxf(mask[n], 0.0f);
  const float zn = (1.0f - al) * agg[n] + al * fv;
  z[n] = zn;
  const float di = dinv[n];
  agg[n] = di * di * zn;
}

// ---------------------------------------------------------------------------
extern "C" void kernel_launch(void* const* d_in, const int* in_sizes, int n_in,
                              void* d_out, int out_size, void* d_ws, size_t ws_size,
                              hipStream_t stream) {
  const float* x     = (const float*)d_in[0];
  const float* mask  = (const float*)d_in[1];
  const int*   ei    = (const int*)  d_in[2];
  const float* W     = (const float*)d_in[3];
  const float* bias  = (const float*)d_in[4];
  const float* alpha = (const float*)d_in[5];

  const int N = in_sizes[1];        // 100000
  const int E = in_sizes[2] / 2;    // 1600000

  // Outputs: z first, then edge_weights
  float* z  = (float*)d_out;
  float* ew = (float*)d_out + N;

  // Workspace carve-up (floats): h | rinv | deg->dinv | enorm | agg
  float* h     = (float*)d_ws;
  float* rinv  = h + (size_t)N * DIM;
  float* dinv  = rinv + N;
  float* enorm = dinv + N;
  float* agg   = enorm + E;

  const int nodeTiles = (N + 15) / 16;
  const int nodeWaves = (N + 7) / 8;
  const int edgeWaves = (E + 7) / 8;
  const int nodeBlks  = (N + 255) / 256;
  const int edgeBlks  = (E + 255) / 256;

  gemm_tanh_wmma<<<nodeTiles, 256, 0, stream>>>(x, W, bias, h, N);
  node_rinv     <<<nodeWaves, 256, 0, stream>>>(h, rinv, dinv, N);
  edge_weights  <<<edgeWaves, 256, 0, stream>>>(h, rinv, ei, ew, E);
  deg_accum     <<<edgeBlks,  256, 0, stream>>>(ei, ew, dinv, E);
  dinv_init     <<<nodeBlks,  256, 0, stream>>>(dinv, mask, z, agg, N);
  edge_norm     <<<edgeBlks,  256, 0, stream>>>(ei, ew, dinv, enorm, E);

  for (int step = 0; step < 5; ++step) {
    appnp_scatter<<<edgeBlks, 256, 0, stream>>>(ei, enorm, z, agg, E);
    appnp_update <<<nodeBlks, 256, 0, stream>>>(mask, dinv, alpha, z, agg, N);
  }
}
